// MultiHeadAttention_31464930410692
// MI455X (gfx1250) — matrix operand order repaired and checked by reading, not compiled, not run on hardware
//
#include <hip/hip_runtime.h>
#include <hip/hip_bf16.h>

// MI455X / gfx1250: wave32, WMMA bf16 path.
// MHA  B=2, S=2048, D=512, H=8, dk=64, faithful to reference bugs:
//   - key = query.view(...)  -> K matrix == Q matrix, Wk/key unused
//   - raw .view head split   -> per-(b,h) slice is a contiguous [S,64] block

#define SEQ    2048
#define DMODEL 512
#define NH     8
#define DKH    64
#define MROWS  (2 * SEQ)          // B*S = 4096

typedef __attribute__((ext_vector_type(16))) __bf16 v16bf;
typedef __attribute__((ext_vector_type(8)))  float  v8f;

static __device__ __forceinline__ unsigned short f2bf(float f) {
  unsigned int u = __builtin_bit_cast(unsigned int, f);
  u += 0x7FFFu + ((u >> 16) & 1u);          // round to nearest even
  return (unsigned short)(u >> 16);
}
static __device__ __forceinline__ __bf16 us2bf(unsigned short s) {
  return __builtin_bit_cast(__bf16, s);
}

// ---------------------------------------------------------------------------
// Projection: Obf[m,n] = (A[m,:] . W[n,:])  fp32 in, bf16 out.
// One wave per 16x64 C-strip: 4 accum tiles, K-loop in steps of 32.
// ---------------------------------------------------------------------------
__global__ __launch_bounds__(256) void proj_gemm_f32_bf16(
    const float* __restrict__ A,            // [4096,512]
    const float* __restrict__ W,            // [512,512] row-major [out,in]
    unsigned short* __restrict__ Obf) {     // [4096,512] bf16
  const int lane = threadIdx.x & 31;
  const int wave = blockIdx.x * (blockDim.x >> 5) + (threadIdx.x >> 5);
  const int m0 = (wave >> 3) * 16;
  const int n0 = (wave & 7) * 64;
  const int lr = lane & 15;
  const int hi = lane >> 4;

  v8f acc[4];
#pragma unroll
  for (int t = 0; t < 4; ++t)
#pragma unroll
    for (int j = 0; j < 8; ++j) acc[t][j] = 0.0f;

  const float* arow = A + (m0 + lr) * DMODEL;
  for (int k0 = 0; k0 < DMODEL; k0 += 32) {
    v16bf af;
    {
      const float* ap = arow + k0;
      const int c0 = hi * 16, c1 = hi * 16 + 8;
#pragma unroll
      for (int j = 0; j < 8; ++j) af[j] = us2bf(f2bf(ap[c0 + j]));
#pragma unroll
      for (int j = 0; j < 8; ++j) af[8 + j] = us2bf(f2bf(ap[c1 + j]));
    }
#pragma unroll
    for (int t = 0; t < 4; ++t) {
      const float* bp = W + (n0 + t * 16 + lr) * DMODEL + k0 + hi * 16;
      v16bf bf;
#pragma unroll
      for (int j = 0; j < 16; ++j) bf[j] = us2bf(f2bf(bp[j]));
      acc[t] = __builtin_amdgcn_wmma_f32_16x16x32_bf16(
          false, af, false, bf, (short)0, acc[t], false, false);
    }
  }
#pragma unroll
  for (int t = 0; t < 4; ++t) {
    const int n = n0 + t * 16 + lr;
#pragma unroll
    for (int r = 0; r < 8; ++r)
      Obf[(m0 + r + 8 * hi) * DMODEL + n] = f2bf(acc[t][r]);
  }
}

// ---------------------------------------------------------------------------
// Fused causal attention per (b,h): Qmat == Kmat (repo bug), flash-style
// streaming softmax. One wave owns 16 query rows; loops key blocks of 32.
// ---------------------------------------------------------------------------
__global__ __launch_bounds__(256) void attn_flash_bf16(
    const unsigned short* __restrict__ Qbf,   // [B,S,D] bf16 (q projection)
    const unsigned short* __restrict__ Vbf,   // [B,S,D] bf16 (v projection)
    unsigned short* __restrict__ Xbf) {       // [B,S,D] bf16 head-concat out
  __shared__ unsigned short sP[8][16 * 32];   // per-wave P staging (C->A transpose)

  const int lane = threadIdx.x & 31;
  const int widx = threadIdx.x >> 5;
  const int wave = blockIdx.x * 8 + widx;     // 0..2047
  const int qb   = wave & 127;                // q block within head
  const int bh   = wave >> 7;                 // 0..15  (b*8+h)
  const int q0   = qb * 16;
  const int b    = bh >> 3, h = bh & 7;
  const unsigned short* Qm = Qbf + bh * (SEQ * DKH);  // contiguous [2048,64]
  const unsigned short* Vm = Vbf + bh * (SEQ * DKH);

  const int lr = lane & 15;
  const int hi = lane >> 4;

  // Q A-fragments for dk chunks [0,32) and [32,64): loaded once.
  v16bf qf[2];
  {
    const unsigned short* qp = Qm + (q0 + lr) * DKH;
#pragma unroll
    for (int c = 0; c < 2; ++c) {
      const int base = c * 32;
#pragma unroll
      for (int j = 0; j < 8; ++j) qf[c][j] = us2bf(qp[base + hi * 16 + j]);
#pragma unroll
      for (int j = 0; j < 8; ++j) qf[c][8 + j] = us2bf(qp[base + hi * 16 + 8 + j]);
    }
  }

  v8f O[4];
  float m_run[8], l_run[8];
#pragma unroll
  for (int t = 0; t < 4; ++t)
#pragma unroll
    for (int j = 0; j < 8; ++j) O[t][j] = 0.0f;
#pragma unroll
  for (int r = 0; r < 8; ++r) { m_run[r] = -3.0e38f; l_run[r] = 0.0f; }

  unsigned short* pl = sP[widx];
  const int kend = q0 + 15;

  for (int kb = 0; kb <= kend; kb += 32) {
    // ---- S = Q Q^T for two 16-key tiles (k-split over dk=64) ----
    v8f sc[2];
#pragma unroll
    for (int t = 0; t < 2; ++t) {
#pragma unroll
      for (int j = 0; j < 8; ++j) sc[t][j] = 0.0f;
#pragma unroll
      for (int c = 0; c < 2; ++c) {
        const unsigned short* kp = Qm + (kb + t * 16 + lr) * DKH + c * 32 + hi * 16;
        v16bf kf;
#pragma unroll
        for (int j = 0; j < 16; ++j) kf[j] = us2bf(kp[j]);
        sc[t] = __builtin_amdgcn_wmma_f32_16x16x32_bf16(
            false, qf[c], false, kf, (short)0, sc[t], false, false);
      }
    }

    // ---- scale, causal mask, streaming softmax update ----
#pragma unroll
    for (int r = 0; r < 8; ++r) {
      const int qrow = q0 + r + 8 * hi;
      float s0 = sc[0][r] * 0.125f;              // 1/sqrt(64)
      float s1 = sc[1][r] * 0.125f;
      if (kb + lr > qrow)      s0 = -1.0e9f;     // reference masked_fill value
      if (kb + 16 + lr > qrow) s1 = -1.0e9f;
      float mx = fmaxf(s0, s1);
      mx = fmaxf(mx, __shfl_xor(mx, 1, 32));
      mx = fmaxf(mx, __shfl_xor(mx, 2, 32));
      mx = fmaxf(mx, __shfl_xor(mx, 4, 32));
      mx = fmaxf(mx, __shfl_xor(mx, 8, 32));
      const float mnew = fmaxf(m_run[r], mx);

      const float p0 = __expf(s0 - mnew);
      const float p1 = __expf(s1 - mnew);
      float rs = p0 + p1;
      rs += __shfl_xor(rs, 1, 32);
      rs += __shfl_xor(rs, 2, 32);
      rs += __shfl_xor(rs, 4, 32);
      rs += __shfl_xor(rs, 8, 32);

      const float alpha = __expf(m_run[r] - mnew);
      l_run[r] = l_run[r] * alpha + rs;
      m_run[r] = mnew;
#pragma unroll
      for (int t = 0; t < 4; ++t) O[t][r] *= alpha;

      const int mrow = r + 8 * hi;               // C-layout row -> LDS row
      pl[mrow * 32 + lr]      = f2bf(p0);
      pl[mrow * 32 + 16 + lr] = f2bf(p1);
    }

    // ---- reload P in A-fragment layout (same-wave LDS ops are in-order) ----
    v16bf pf;
    {
      const unsigned short* pp = pl + lr * 32;
#pragma unroll
      for (int j = 0; j < 8; ++j) pf[j] = us2bf(pp[hi * 16 + j]);
#pragma unroll
      for (int j = 0; j < 8; ++j) pf[8 + j] = us2bf(pp[hi * 16 + 8 + j]);
    }

    // ---- O += P @ V(block) ----
#pragma unroll
    for (int t = 0; t < 4; ++t) {
      const unsigned short* vp = Vm + (kb + hi * 16) * DKH + t * 16 + lr;
      v16bf vf;
#pragma unroll
      for (int j = 0; j < 16; ++j) vf[j] = us2bf(vp[j * DKH]);
      O[t] = __builtin_amdgcn_wmma_f32_16x16x32_bf16(
          false, pf, false, vf, (short)0, O[t], false, false);
    }
  }

  // ---- epilogue: normalize and write head-concat layout [B,S,H*dk] ----
#pragma unroll
  for (int t = 0; t < 4; ++t) {
#pragma unroll
    for (int r = 0; r < 8; ++r) {
      const int qrow = q0 + r + 8 * hi;
      const float val = O[t][r] / l_run[r];
      Xbf[(b * SEQ + qrow) * DMODEL + h * DKH + t * 16 + lr] = f2bf(val);
    }
  }
}

// ---------------------------------------------------------------------------
// Output projection: Out[m,n] = x[m,:] . Wo[n,:]  (bf16 A, fp32 W, fp32 out)
// ---------------------------------------------------------------------------
__global__ __launch_bounds__(256) void out_gemm_bf16_f32(
    const unsigned short* __restrict__ Abf,  // [4096,512] bf16
    const float* __restrict__ W,             // Wo [512,512]
    float* __restrict__ Out) {               // [4096,512] fp32
  const int lane = threadIdx.x & 31;
  const int wave = blockIdx.x * (blockDim.x >> 5) + (threadIdx.x >> 5);
  const int m0 = (wave >> 3) * 16;
  const int n0 = (wave & 7) * 64;
  const int lr = lane & 15;
  const int hi = lane >> 4;

  v8f acc[4];
#pragma unroll
  for (int t = 0; t < 4; ++t)
#pragma unroll
    for (int j = 0; j < 8; ++j) acc[t][j] = 0.0f;

  const unsigned short* arow = Abf + (m0 + lr) * DMODEL;
  for (int k0 = 0; k0 < DMODEL; k0 += 32) {
    v16bf af;
    {
      const unsigned short* ap = arow + k0;
      const int c0 = hi * 16, c1 = hi * 16 + 8;
#pragma unroll
      for (int j = 0; j < 8; ++j) af[j] = us2bf(ap[c0 + j]);
#pragma unroll
      for (int j = 0; j < 8; ++j) af[8 + j] = us2bf(ap[c1 + j]);
    }
#pragma unroll
    for (int t = 0; t < 4; ++t) {
      const float* bp = W + (n0 + t * 16 + lr) * DMODEL + k0 + hi * 16;
      v16bf bf;
#pragma unroll
      for (int j = 0; j < 16; ++j) bf[j] = us2bf(f2bf(bp[j]));
      acc[t] = __builtin_amdgcn_wmma_f32_16x16x32_bf16(
          false, af, false, bf, (short)0, acc[t], false, false);
    }
  }
#pragma unroll
  for (int t = 0; t < 4; ++t) {
    const int n = n0 + t * 16 + lr;
#pragma unroll
    for (int r = 0; r < 8; ++r)
      Out[(m0 + r + 8 * hi) * DMODEL + n] = acc[t][r];
  }
}

// ---------------------------------------------------------------------------
extern "C" void kernel_launch(void* const* d_in, const int* in_sizes, int n_in,
                              void* d_out, int out_size, void* d_ws, size_t ws_size,
                              hipStream_t stream) {
  (void)in_sizes; (void)n_in; (void)out_size; (void)ws_size;
  const float* query = (const float*)d_in[0];
  // d_in[1] = key   : unused (reference discards key projection)
  const float* value = (const float*)d_in[2];
  // d_in[3] = mask  : causal tril, hard-coded in attention kernel
  const float* Wq = (const float*)d_in[4];
  // d_in[5] = Wk    : unused
  const float* Wv = (const float*)d_in[6];
  const float* Wo = (const float*)d_in[7];
  float* out = (float*)d_out;

  unsigned short* qbf = (unsigned short*)d_ws;                 // 4 MB
  unsigned short* vbf = qbf + (size_t)MROWS * DMODEL;          // 4 MB
  unsigned short* xbf = vbf + (size_t)MROWS * DMODEL;          // 4 MB

  // 2048 waves per GEMM / attention pass; 8 waves (256 threads) per block.
  proj_gemm_f32_bf16<<<256, 256, 0, stream>>>(query, Wq, qbf);
  proj_gemm_f32_bf16<<<256, 256, 0, stream>>>(value, Wv, vbf);
  attn_flash_bf16<<<256, 256, 0, stream>>>(qbf, vbf, xbf);
  out_gemm_bf16_f32<<<256, 256, 0, stream>>>(xbf, Wo, out);
}
